// CausalSelfAttention_5394478924039
// MI455X (gfx1250) — matrix operand order repaired and checked by reading, not compiled, not run on hardware
//
#include <hip/hip_runtime.h>

typedef __attribute__((ext_vector_type(16))) __bf16 v16bf;
typedef __attribute__((ext_vector_type(8)))  __bf16 v8bf;
typedef __attribute__((ext_vector_type(8)))  float  v8f;

#define T_SEQ 2048
#define NHEAD 16
#define HDIM  64

// ---------------------------------------------------------------------------
// WMMA bf16 16x16x32 fragment maps (wave32):
// A (16x32, MxK): lane L -> M = L%16 ; g = L>>4
//   elem e=0..7  -> K = g*8 + e ; elem e=8..15 -> K = 16 + g*8 + (e-8)
// B (32x16, KxN): lane L -> N = L%16 ; elem e -> K = (L>>4)*8 + (e&7) + (e>>3)*16
// C/D (16x16):    lane L -> N = L%16 ; vgpr r -> M = r + 8*(L>>4)
// ---------------------------------------------------------------------------

// Repack fp32 W[K][N] into fragment-native bf16: [(nt*K/32 + kt)*32 + lane][16]
__global__ __launch_bounds__(256) void pack_w_bf16(
    const float* __restrict__ W, __bf16* __restrict__ Wp, int K, int N)
{
    const int tid = blockIdx.x * 256 + threadIdx.x;   // packed linear index
    const int e    = tid & 15;
    const int lane = (tid >> 4) & 31;
    const int KT   = K >> 5;
    const int kt   = (tid >> 9) % KT;
    const int nt   = (tid >> 9) / KT;
    const int k = kt * 32 + (lane >> 4) * 8 + (e & 7) + (e >> 3) * 16;
    const int n = nt * 16 + (lane & 15);
    Wp[tid] = (__bf16)W[(size_t)k * N + n];
}

__global__ __launch_bounds__(256) void cvt_f32_bf16(
    const float* __restrict__ X, __bf16* __restrict__ Xb)
{
    const int tid = blockIdx.x * 256 + threadIdx.x;
    Xb[tid] = (__bf16)X[tid];
}

__device__ inline v16bf ld_a_frag(const __bf16* p) {
    // p points at row base + g*8; chunks at +0 and +16 elements
    const v8bf* q = (const v8bf*)p;
    v16bf a;
#pragma unroll
    for (int i = 0; i < 8; ++i) { a[i] = q[0][i]; a[i + 8] = q[2][i]; }
    return a;
}

// C = A(MxK bf16 row-major) * Wp(packed bf16) + bias, f32 out.
// 256 thr = 8 waves as 2(M) x 4(N); wave computes 16x64 (4 wmma / k-step).
// Ping-pong double buffering with pointer+immediate addressing; tail peeled.
__global__ __launch_bounds__(256) void gemm_bf16_wmma(
    const __bf16* __restrict__ A, const __bf16* __restrict__ Wp,
    const float* __restrict__ bias, float* __restrict__ C,
    int M, int N, int K)
{
    const int lane = threadIdx.x & 31;
    const int wave = threadIdx.x >> 5;
    const int g    = lane >> 4;
    const int ln   = lane & 15;
    const int m0 = blockIdx.y * 32  + (wave >> 2) * 16;
    const int n0 = blockIdx.x * 256 + (wave & 3) * 64;
    const int KT  = K >> 5;
    const int nt0 = n0 >> 4;

    // running pointers (strength-reduced addressing)
    const __bf16* pA = A + (size_t)(m0 + ln) * K + g * 8;
    const __bf16* pB[4];
#pragma unroll
    for (int t = 0; t < 4; ++t)
        pB[t] = Wp + ((size_t)(nt0 + t) * KT * 32 + lane) * 16;

    v8f c[4];
#pragma unroll
    for (int t = 0; t < 4; ++t)
#pragma unroll
        for (int r = 0; r < 8; ++r) c[t][r] = 0.0f;

    // prime slot 0 with kt = 0
    v16bf a0 = ld_a_frag(pA);
    v16bf b0[4];
#pragma unroll
    for (int t = 0; t < 4; ++t) b0[t] = *(const v16bf*)pB[t];

    int kt = 0;
    for (; kt + 2 < KT; kt += 2) {
        // preload slot 1 (kt+1) at fixed immediate offsets
        v16bf a1 = ld_a_frag(pA + 32);
        v16bf b1[4];
#pragma unroll
        for (int t = 0; t < 4; ++t) b1[t] = *(const v16bf*)(pB[t] + 512);
#pragma unroll
        for (int t = 0; t < 4; ++t)
            c[t] = __builtin_amdgcn_wmma_f32_16x16x32_bf16(
                false, a0, false, b0[t], (short)0, c[t], false, false);

        // advance and preload slot 0 (kt+2)
        pA += 64;
        a0 = ld_a_frag(pA);
#pragma unroll
        for (int t = 0; t < 4; ++t) { pB[t] += 1024; b0[t] = *(const v16bf*)pB[t]; }
#pragma unroll
        for (int t = 0; t < 4; ++t)
            c[t] = __builtin_amdgcn_wmma_f32_16x16x32_bf16(
                false, a1, false, b1[t], (short)0, c[t], false, false);
    }
    // peeled tail pair (KT-2, KT-1): no further preload
    {
        v16bf a1 = ld_a_frag(pA + 32);
        v16bf b1[4];
#pragma unroll
        for (int t = 0; t < 4; ++t) b1[t] = *(const v16bf*)(pB[t] + 512);
#pragma unroll
        for (int t = 0; t < 4; ++t)
            c[t] = __builtin_amdgcn_wmma_f32_16x16x32_bf16(
                false, a0, false, b0[t], (short)0, c[t], false, false);
#pragma unroll
        for (int t = 0; t < 4; ++t)
            c[t] = __builtin_amdgcn_wmma_f32_16x16x32_bf16(
                false, a1, false, b1[t], (short)0, c[t], false, false);
    }

#pragma unroll
    for (int t = 0; t < 4; ++t) {
        const float bv = bias[n0 + t * 16 + ln];
#pragma unroll
        for (int r = 0; r < 8; ++r) {
            const size_t rr = (size_t)(m0 + r + 8 * g) * N;
            C[rr + n0 + t * 16 + ln] = c[t][r] + bv;
        }
    }
}

// ---------------------------------------------------------------------------
// RoPE (reference flavor) + split. Q (x0.125) and K as bf16 [B,H,T,D];
// V stored TRANSPOSED as bf16 [B,H,D,T] so PV B-fragments are contiguous.
// ---------------------------------------------------------------------------
__global__ __launch_bounds__(256) void rope_split(
    const float* __restrict__ qkv,
    __bf16* __restrict__ Qb, __bf16* __restrict__ Kb, __bf16* __restrict__ Vt)
{
    const int tid = blockIdx.x * blockDim.x + threadIdx.x;
    const int dp = tid & 31;
    const int h  = (tid >> 5) & 15;
    const int t  = (tid >> 9) & (T_SEQ - 1);
    const int b  = tid >> 20;
    const int d0 = dp * 2;

    const size_t src = ((size_t)(b * T_SEQ + t)) * 3072 + h * HDIM;
    const float q0v = qkv[src + d0],        q1v = qkv[src + d0 + 1];
    const float k0v = qkv[src + 1024 + d0], k1v = qkv[src + 1024 + d0 + 1];
    const float v0v = qkv[src + 2048 + d0], v1v = qkv[src + 2048 + d0 + 1];

    const float LN1E4_32 = 9.2103403719761836f / 32.0f;
    const float tf = (float)t;
    const float a0 = tf * __expf(-(float)(d0 & 31) * LN1E4_32);
    const float a1 = tf * __expf(-(float)((d0 + 1) & 31) * LN1E4_32);
    const float c0 = cosf(a0), s0 = sinf(a0);
    const float c1 = cosf(a1), s1 = sinf(a1);

    const float qe = q0v * c0 - q1v * s0;
    const float qo = q1v * c1 + q0v * s1;
    const float ke = k0v * c0 - k1v * s0;
    const float ko = k1v * c1 + k0v * s1;

    const int bh = b * NHEAD + h;
    const size_t dst = ((size_t)bh * T_SEQ + t) * HDIM + d0;
    const float sc = 0.125f;
    Qb[dst]     = (__bf16)(qe * sc);
    Qb[dst + 1] = (__bf16)(qo * sc);
    Kb[dst]     = (__bf16)ke;
    Kb[dst + 1] = (__bf16)ko;
    const size_t vdst = ((size_t)bh * HDIM + d0) * T_SEQ + t;
    Vt[vdst]         = (__bf16)v0v;
    Vt[vdst + T_SEQ] = (__bf16)v1v;
}

// ---------------------------------------------------------------------------
// Flash attention: one 16-query tile per wave, 32-key tiles, online softmax.
// Pointer+immediate addressing for all K/V fragment loads; softmax VALU
// overlaps in-flight V loads and co-executes with the XDL wmmas.
// ---------------------------------------------------------------------------
__global__ __launch_bounds__(256) void flash_attn(
    const __bf16* __restrict__ Qb, const __bf16* __restrict__ Kb,
    const __bf16* __restrict__ Vt, __bf16* __restrict__ Yb)
{
    __shared__ __bf16 Pl[8][16 * 32];

    const int lane = threadIdx.x & 31;
    const int wave = threadIdx.x >> 5;
    const int g    = lane >> 4;
    const int ln   = lane & 15;

    const int qt = blockIdx.x * 8 + wave;
    const int qi = qt & (T_SEQ / 16 - 1);
    const int bh = qt >> 7;
    const int q0 = qi * 16;

    const __bf16* Q = Qb + (size_t)bh * T_SEQ * HDIM;
    // running fragment pointers (advance per key tile)
    const __bf16* pK = Kb + (size_t)bh * T_SEQ * HDIM + (size_t)ln * HDIM + g * 16;
    const __bf16* pV = Vt + (size_t)bh * HDIM * T_SEQ + (size_t)ln * T_SEQ + g * 16;

    v16bf qa[2];
    {
        const __bf16* p = Q + (size_t)(q0 + ln) * HDIM + g * 16;
        qa[0] = *(const v16bf*)p;
        qa[1] = *(const v16bf*)(p + 32);
    }

    float m[8], l[8];
    v8f o[4];
#pragma unroll
    for (int r = 0; r < 8; ++r) { m[r] = -1e30f; l[r] = 0.0f; }
#pragma unroll
    for (int t = 0; t < 4; ++t)
#pragma unroll
        for (int r = 0; r < 8; ++r) o[t][r] = 0.0f;

    __bf16* P = &Pl[wave][0];
    const int jend = q0 + 16;

    for (int j0 = 0; j0 < jend; j0 += 32) {
        // --- all fragment loads at immediate offsets from pK/pV ---
        v16bf kb[4];   // [half*2 + ks]
#pragma unroll
        for (int half = 0; half < 2; ++half)
#pragma unroll
            for (int ks = 0; ks < 2; ++ks)
                kb[half * 2 + ks] =
                    *(const v16bf*)(pK + half * 16 * HDIM + ks * 32);
        v16bf vb[4];
#pragma unroll
        for (int t = 0; t < 4; ++t)
            vb[t] = *(const v16bf*)(pV + (size_t)t * 16 * T_SEQ);
        pK += 32 * HDIM;
        pV += 32;

        // --- S = Q K^T ---
        v8f s[2];
#pragma unroll
        for (int half = 0; half < 2; ++half) {
#pragma unroll
            for (int r = 0; r < 8; ++r) s[half][r] = 0.0f;
#pragma unroll
            for (int ks = 0; ks < 2; ++ks)
                s[half] = __builtin_amdgcn_wmma_f32_16x16x32_bf16(
                    false, qa[ks], false, kb[half * 2 + ks],
                    (short)0, s[half], false, false);
        }

        if (j0 + 31 > q0) {
#pragma unroll
            for (int half = 0; half < 2; ++half) {
                const int j = j0 + half * 16 + ln;
#pragma unroll
                for (int r = 0; r < 8; ++r) {
                    if (j > q0 + r + 8 * g) s[half][r] = -1e30f;
                }
            }
        }

        // --- online softmax (overlaps in-flight V loads) ---
        float alpha[8];
#pragma unroll
        for (int r = 0; r < 8; ++r) {
            float v = fmaxf(s[0][r], s[1][r]);
#pragma unroll
            for (int off = 8; off > 0; off >>= 1)
                v = fmaxf(v, __shfl_xor(v, off, 16));
            const float mn = fmaxf(m[r], v);
            alpha[r] = __expf(m[r] - mn);
            m[r] = mn;
            const float p0 = __expf(s[0][r] - mn);
            const float p1 = __expf(s[1][r] - mn);
            s[0][r] = p0; s[1][r] = p1;
            float rs = p0 + p1;
#pragma unroll
            for (int off = 8; off > 0; off >>= 1)
                rs += __shfl_xor(rs, off, 16);
            l[r] = l[r] * alpha[r] + rs;
        }
#pragma unroll
        for (int t = 0; t < 4; ++t)
#pragma unroll
            for (int r = 0; r < 8; ++r) o[t][r] *= alpha[r];

        // --- P: C-layout -> LDS -> A-layout (per-wave; DS ops in-order) ---
#pragma unroll
        for (int half = 0; half < 2; ++half)
#pragma unroll
            for (int r = 0; r < 8; ++r)
                P[(r + 8 * g) * 32 + half * 16 + ln] = (__bf16)s[half][r];

        v16bf pa = *(const v16bf*)(P + ln * 32 + g * 16);

        // --- O += P * V ---
#pragma unroll
        for (int t = 0; t < 4; ++t)
            o[t] = __builtin_amdgcn_wmma_f32_16x16x32_bf16(
                false, pa, false, vb[t], (short)0, o[t], false, false);
    }

    const int b = bh >> 4, h = bh & 15;
#pragma unroll
    for (int r = 0; r < 8; ++r) {
        const float inv = 1.0f / l[r];
        const int qrow = q0 + r + 8 * g;
        const size_t base = ((size_t)(b * T_SEQ + qrow)) * 1024 + h * HDIM + ln;
#pragma unroll
        for (int t = 0; t < 4; ++t)
            Yb[base + t * 16] = (__bf16)(o[t][r] * inv);
    }
}

// ---------------------------------------------------------------------------
extern "C" void kernel_launch(void* const* d_in, const int* in_sizes, int n_in,
                              void* d_out, int out_size, void* d_ws, size_t ws_size,
                              hipStream_t stream) {
    (void)in_sizes; (void)n_in; (void)out_size; (void)ws_size;
    const float* x      = (const float*)d_in[0];
    const float* w_attn = (const float*)d_in[1];
    const float* b_attn = (const float*)d_in[2];
    const float* w_proj = (const float*)d_in[3];
    const float* b_proj = (const float*)d_in[4];
    float* out = (float*)d_out;

    char* ws = (char*)d_ws;
    const size_t MB = 1024 * 1024;
    float*  qkv = (float*)ws;                          // 48 MB fp32
    __bf16* Qb  = (__bf16*)(ws + 48 * MB);             //  8 MB
    __bf16* Kb  = (__bf16*)(ws + 56 * MB);             //  8 MB
    __bf16* Vt  = (__bf16*)(ws + 64 * MB);             //  8 MB (transposed)
    __bf16* Yb  = (__bf16*)(ws + 72 * MB);             //  8 MB
    __bf16* Xb  = (__bf16*)(ws + 80 * MB);             //  8 MB
    __bf16* WaP = (__bf16*)(ws + 88 * MB);             //  6 MB packed w_attn
    __bf16* WpP = (__bf16*)(ws + 94 * MB);             //  2 MB packed w_proj

    const int M = 2 * T_SEQ;  // 4096

    // 0) one-time bf16 conversions / fragment-native repacks
    pack_w_bf16<<<(1024 * 3072) / 256, 256, 0, stream>>>(w_attn, WaP, 1024, 3072);
    pack_w_bf16<<<(1024 * 1024) / 256, 256, 0, stream>>>(w_proj, WpP, 1024, 1024);
    cvt_f32_bf16<<<(M * 1024) / 256, 256, 0, stream>>>(x, Xb);
    // 1) QKV projection
    gemm_bf16_wmma<<<dim3(3072 / 256, M / 32), 256, 0, stream>>>(
        Xb, WaP, b_attn, qkv, M, 3072, 1024);
    // 2) RoPE + split (Q pre-scaled by 1/sqrt(D)); V transposed
    rope_split<<<(2u << 20) / 256, 256, 0, stream>>>(qkv, Qb, Kb, Vt);
    // 3) Flash attention -> bf16 activations
    flash_attn<<<512, 256, 0, stream>>>(Qb, Kb, Vt, Yb);
    // 4) Output projection
    gemm_bf16_wmma<<<dim3(1024 / 256, M / 32), 256, 0, stream>>>(
        Yb, WpP, b_proj, out, M, 1024, 1024);
}
